// LearnableCentroidsWrapper_45543833207260
// MI455X (gfx1250) — matrix operand order repaired and checked by reading, not compiled, not run on hardware
//
#include <hip/hip_runtime.h>

// ---------------------------------------------------------------------------
// LearnableCentroids assignment for gfx1250 (MI455X, CDNA5, wave32).
//  Phase A (tiny): merge centroids via threshold-graph connected components,
//                  then split merged centroids into bf16 hi/lo planes.
//  Phase B (big) : argmin_c (m2[c] - 2*X@merged^T) as split-precision bf16
//                  GEMM: S = Ah*Bh + Ah*Bl + Al*Bh  (v_wmma_f32_16x16x32_bf16,
//                  XDL path), with double-buffered async global->LDS staging.
// ---------------------------------------------------------------------------

typedef float  v8f   __attribute__((ext_vector_type(8)));
typedef __bf16 v16bf __attribute__((ext_vector_type(16)));

#define D 512            // feature dim
#define K 512            // num centroids
#define BSTRIDE_US 40    // LDS row stride in bf16 (32 data + 8 pad = 20 dwords)

// ------------------------- helpers -----------------------------------------

__device__ __forceinline__ unsigned short bf16_rne(float f) {
  unsigned u = __float_as_uint(f);
  unsigned r = u + 0x7FFFu + ((u >> 16) & 1u);   // round-to-nearest-even
  return (unsigned short)(r >> 16);
}

struct B32B { uint4 a, b; };
__device__ __forceinline__ v16bf lds_b16x16(const unsigned short* p) {
  B32B t;
  t.a = *(const uint4*)p;         // 16B aligned (row stride 80B, col offset 32B)
  t.b = *(const uint4*)(p + 8);
  return __builtin_bit_cast(v16bf, t);
}

// gfx1250 async copy: global -> LDS, tracked by ASYNCcnt (no VGPR data path).
__device__ __forceinline__ void async_copy_b128(unsigned lds_off, const void* g) {
  asm volatile("global_load_async_to_lds_b128 %0, %1, off"
               :: "v"(lds_off), "v"((unsigned long long)(uintptr_t)g)
               : "memory");
}
__device__ __forceinline__ void wait_async0() {          // all copies done
  asm volatile("s_wait_asynccnt 0" ::: "memory");
}
__device__ __forceinline__ void wait_async4() {          // older group done,
  asm volatile("s_wait_asynccnt 4" ::: "memory");        // newest 4 in flight
}

// ---------------- Phase A kernels (all tiny) -------------------------------

__global__ void k_c2(const float* __restrict__ C, float* __restrict__ c2,
                     unsigned* __restrict__ maxbits) {
  __shared__ float s[256];
  const int i = blockIdx.x, tid = threadIdx.x;
  if (i == 0 && tid == 0) *maxbits = 0u;
  float v = 0.f;
  for (int d = tid; d < D; d += 256) { float x = C[i * D + d]; v += x * x; }
  s[tid] = v; __syncthreads();
  for (int off = 128; off > 0; off >>= 1) {
    if (tid < off) s[tid] += s[tid + off];
    __syncthreads();
  }
  if (tid == 0) c2[i] = s[0];
}

__global__ void k_dmax(const float* __restrict__ C, const float* __restrict__ c2,
                       unsigned* __restrict__ maxbits) {
  __shared__ float s[256];
  const int tid = threadIdx.x;
  const int gid = blockIdx.x * 256 + tid;         // 0 .. K*K-1
  const int i = gid >> 9, j = gid & (K - 1);
  float dot = 0.f;
  for (int d = 0; d < D; ++d) dot += C[i * D + d] * C[j * D + d];
  float d2 = fmaxf(c2[i] + c2[j] - 2.f * dot, 0.f);
  s[tid] = sqrtf(d2); __syncthreads();
  for (int off = 128; off > 0; off >>= 1) {
    if (tid < off) s[tid] = fmaxf(s[tid], s[tid + off]);
    __syncthreads();
  }
  if (tid == 0) atomicMax(maxbits, __float_as_uint(s[0]));  // values >= 0
}

__global__ void k_adj(const float* __restrict__ C, const float* __restrict__ c2,
                      const unsigned* __restrict__ maxbits, unsigned* __restrict__ adj) {
  const int gid = blockIdx.x * 256 + threadIdx.x;  // 0 .. K*16-1
  const int i = gid >> 4, w = gid & 15;
  const float thr = 0.1f * __uint_as_float(*maxbits);
  unsigned bits = 0u;
  for (int b = 0; b < 32; ++b) {
    const int j = w * 32 + b;
    float dot = 0.f;
    for (int d = 0; d < D; ++d) dot += C[i * D + d] * C[j * D + d];
    float dist = sqrtf(fmaxf(c2[i] + c2[j] - 2.f * dot, 0.f));
    if (dist < thr) bits |= (1u << b);
  }
  adj[i * 16 + w] = bits;
}

__global__ void k_closure(const unsigned* __restrict__ adj, int* __restrict__ labels) {
  __shared__ unsigned R[K][16];
  const int i = threadIdx.x;                       // 512 threads, one row each
  unsigned my[16];
  for (int w = 0; w < 16; ++w) { my[w] = adj[i * 16 + w]; R[i][w] = my[w]; }
  __syncthreads();
  for (int it = 0; it < 10; ++it) {                // ceil(log2(512)) + 1
    unsigned acc[16];
    for (int w = 0; w < 16; ++w) acc[w] = my[w];
    for (int w = 0; w < 16; ++w) {
      unsigned bits = my[w];
      while (bits) {
        const int j = w * 32 + __builtin_ctz(bits);
        bits &= bits - 1;
        for (int u = 0; u < 16; ++u) acc[u] |= R[j][u];
      }
    }
    __syncthreads();
    for (int w = 0; w < 16; ++w) { my[w] = acc[w]; R[i][w] = acc[w]; }
    __syncthreads();
  }
  int lab = 0;
  for (int w = 0; w < 16; ++w) {
    if (my[w]) { lab = w * 32 + __builtin_ctz(my[w]); break; }
  }
  labels[i] = lab;
}

__global__ void k_compact(const int* __restrict__ labels, int* __restrict__ compact) {
  __shared__ int rf[K];
  const int i = threadIdx.x;
  const int lab = labels[i];
  rf[i] = (lab == i) ? 1 : 0;
  __syncthreads();
  int c = 0;
  for (int r = 0; r < lab; ++r) c += rf[r];
  compact[i] = c;
}

__global__ void k_merge(const float* __restrict__ C, const int* __restrict__ compact,
                        float* __restrict__ merged) {
  __shared__ int sc[K];
  const int tid = threadIdx.x;
  for (int i = tid; i < K; i += 256) sc[i] = compact[i];
  __syncthreads();
  const int gid = blockIdx.x * 256 + tid;          // 0 .. K*D-1
  const int c = gid >> 9, d = gid & (D - 1);
  float sum = 0.f; int cnt = 0;
  for (int i = 0; i < K; ++i) {
    if (sc[i] == c) { sum += C[i * D + d]; ++cnt; }
  }
  merged[gid] = (cnt > 0) ? (sum / (float)cnt) : 1e9f;
}

__global__ void k_m2(const float* __restrict__ merged, float* __restrict__ m2) {
  __shared__ float s[256];
  const int c = blockIdx.x, tid = threadIdx.x;
  float v = 0.f;
  for (int d = tid; d < D; d += 256) { float x = merged[c * D + d]; v += x * x; }
  s[tid] = v; __syncthreads();
  for (int off = 128; off > 0; off >>= 1) {
    if (tid < off) s[tid] += s[tid + off];
    __syncthreads();
  }
  if (tid == 0) m2[c] = s[0];
}

// Split merged fp32 into bf16 hi (truncation) / lo (RNE of remainder).
__global__ void k_bsplit(const float* __restrict__ merged,
                         unsigned short* __restrict__ mh,
                         unsigned short* __restrict__ ml) {
  const int gid = blockIdx.x * 256 + threadIdx.x;  // 0 .. K*D-1
  const float x = merged[gid];
  const unsigned u = __float_as_uint(x);
  mh[gid] = (unsigned short)(u >> 16);             // hi = truncate
  const float fhi = __uint_as_float(u & 0xFFFF0000u);
  ml[gid] = bf16_rne(x - fhi);                     // lo = RNE(x - hi)
}

// ---------------- Phase B: split-bf16 WMMA GEMM + fused argmin -------------
// Block = 256 threads = 8 waves; each wave owns one 16-row M strip (M=128/blk).
// N looped in 4 slabs of 128 (8 tiles of 16x16 per wave, acc = 64 VGPRs).
// K loop: 16 chunks of 32; B chunk async-staged to double-buffered LDS so the
// L2->LDS copy of chunk c+1 overlaps the WMMA work of chunk c.
__global__ __launch_bounds__(256) void k_assign(const float* __restrict__ X,
                                                const unsigned short* __restrict__ mh,
                                                const unsigned short* __restrict__ ml,
                                                const float* __restrict__ m2,
                                                int* __restrict__ out) {
  __shared__ __align__(16) unsigned short BsHi[2][128 * BSTRIDE_US];  // 2x10 KB
  __shared__ __align__(16) unsigned short BsLo[2][128 * BSTRIDE_US];  // 2x10 KB
  __shared__ float m2s[K];

  const int tid  = threadIdx.x;
  const int lane = tid & 31;
  const int wave = tid >> 5;
  const int h    = lane >> 4;                 // K-half of WMMA operand layout
  const int l16  = lane & 15;
  const int mrow = blockIdx.x * 128 + wave * 16 + l16;

  for (int i = tid; i < K; i += 256) m2s[i] = m2[i];

  float bv[8]; int bc[8];
#pragma unroll
  for (int v = 0; v < 8; ++v) { bv[v] = 3.0e38f; bc[v] = 0; }

  const float* Arow = X + (size_t)mrow * D;

  // Per-thread async staging map: 512 b128-chunks per plane, 2 per thread.
  const int sc0 = tid >> 2,         sq0 = (tid & 3) * 8;
  const int sc1 = (tid + 256) >> 2, sq1 = (tid & 3) * 8;
  unsigned dh[2][2], dl[2][2];
#pragma unroll
  for (int p = 0; p < 2; ++p) {
    dh[p][0] = (unsigned)(uintptr_t)&BsHi[p][sc0 * BSTRIDE_US + sq0];
    dh[p][1] = (unsigned)(uintptr_t)&BsHi[p][sc1 * BSTRIDE_US + sq1];
    dl[p][0] = (unsigned)(uintptr_t)&BsLo[p][sc0 * BSTRIDE_US + sq0];
    dl[p][1] = (unsigned)(uintptr_t)&BsLo[p][sc1 * BSTRIDE_US + sq1];
  }

  // Issue one chunk's 4 async b128 copies (hi+lo planes) into buffer p.
  auto issue_group = [&](int cbase, int k0, int p) {
    const size_t g0 = (size_t)(cbase + sc0) * D + k0 + sq0;
    const size_t g1 = (size_t)(cbase + sc1) * D + k0 + sq1;
    async_copy_b128(dh[p][0], mh + g0);
    async_copy_b128(dh[p][1], mh + g1);
    async_copy_b128(dl[p][0], ml + g0);
    async_copy_b128(dl[p][1], ml + g1);
  };

  const v8f zero = {0.f, 0.f, 0.f, 0.f, 0.f, 0.f, 0.f, 0.f};

  issue_group(0, 0, 0);                       // prologue: (slab0, chunk0) -> buf0

  for (int slab = 0; slab < 4; ++slab) {
    const int cbase = slab * 128;
    v8f acc[8];
#pragma unroll
    for (int t = 0; t < 8; ++t) acc[t] = zero;

    for (int chunk = 0; chunk < K / 32; ++chunk) {
      const int k0 = chunk * 32;
      const int p  = chunk & 1;               // 16 chunks/slab: parity continuous

      // Prefetch next (slab,chunk) into the other buffer; retire older group.
      if (!(slab == 3 && chunk == 15)) {
        const int ns = (chunk == 15) ? slab + 1 : slab;
        const int nk = (chunk == 15) ? 0 : (chunk + 1) * 32;
        issue_group(ns * 128, nk, p ^ 1);
        wait_async4();                        // buf p's copies complete
      } else {
        wait_async0();
      }
      __syncthreads();                        // everyone's buf p writes visible

      // A operand (this code's contiguous k order): lane half h holds K = 16h+{0..15}
      // (matches B's LDS read); load as fp32 and split into bf16 hi/lo on the fly.
      const float* ap = Arow + k0 + 16 * h;
      float xs[16];
      *(float4*)&xs[0]  = *(const float4*)(ap);
      *(float4*)&xs[4]  = *(const float4*)(ap + 4);
      *(float4*)&xs[8]  = *(const float4*)(ap + 8);
      *(float4*)&xs[12] = *(const float4*)(ap + 12);
      v16bf ah, al;
#pragma unroll
      for (int j = 0; j < 16; ++j) {
        const unsigned u = __float_as_uint(xs[j]);
        ah[j] = __builtin_bit_cast(__bf16, (unsigned short)(u >> 16));
        const float fhi = __uint_as_float(u & 0xFFFF0000u);
        al[j] = __builtin_bit_cast(__bf16, bf16_rne(xs[j] - fhi));
      }

#pragma unroll
      for (int t = 0; t < 8; ++t) {
        const int cr = (t * 16 + l16) * BSTRIDE_US + 16 * h;  // B col=centroid
        const v16bf bh = lds_b16x16(&BsHi[p][cr]);
        const v16bf bl = lds_b16x16(&BsLo[p][cr]);
        acc[t] = __builtin_amdgcn_wmma_f32_16x16x32_bf16(
            false, ah, false, bh, (short)0, acc[t], false, false);
        acc[t] = __builtin_amdgcn_wmma_f32_16x16x32_bf16(
            false, al, false, bh, (short)0, acc[t], false, false);
        acc[t] = __builtin_amdgcn_wmma_f32_16x16x32_bf16(
            false, ah, false, bl, (short)0, acc[t], false, false);
      }
      __syncthreads();    // compute done before next iter overwrites buf p
    }

    // slab epilogue: score = m2[c] - 2*S; lane holds row M = v + 8h per tile.
#pragma unroll
    for (int t = 0; t < 8; ++t) {
      const int c = cbase + t * 16 + l16;
      const float mm = m2s[c];
#pragma unroll
      for (int v = 0; v < 8; ++v) {
        const float s = mm - 2.0f * acc[t][v];
        if (s < bv[v]) { bv[v] = s; bc[v] = c; }   // ties -> smaller c (asc scan)
      }
    }
  }

  // Reduce across the 16 lanes of each half (masks <= 8 keep halves separate).
#pragma unroll
  for (int m = 1; m <= 8; m <<= 1) {
#pragma unroll
    for (int v = 0; v < 8; ++v) {
      const float ov = __shfl_xor(bv[v], m, 32);
      const int   oc = __shfl_xor(bc[v], m, 32);
      if (ov < bv[v] || (ov == bv[v] && oc < bc[v])) { bv[v] = ov; bc[v] = oc; }
    }
  }
  if (l16 == 0) {                               // lanes 0 and 16: 8 rows each
#pragma unroll
    for (int v = 0; v < 8; ++v) {
      out[blockIdx.x * 128 + wave * 16 + h * 8 + v] = bc[v];
    }
  }
}

// ---------------------------------------------------------------------------

extern "C" void kernel_launch(void* const* d_in, const int* in_sizes, int n_in,
                              void* d_out, int out_size, void* d_ws, size_t ws_size,
                              hipStream_t stream) {
  const float* X = (const float*)d_in[0];       // [N, 512]
  const float* C = (const float*)d_in[1];       // [512, 512]
  const int N = in_sizes[0] / D;                // 262144

  // Workspace layout (~2.05 MB)
  char* w = (char*)d_ws;
  float*          merged = (float*)w;                                // 1 MB
  unsigned short* mh     = (unsigned short*)(w + (1 << 20));         // 512 KB
  unsigned short* ml     = (unsigned short*)(w + (1 << 20) + (512 << 10)); // 512 KB
  char* misc = w + (2 << 20);
  float*    c2      = (float*)misc;
  unsigned* maxbits = (unsigned*)(misc + 2048);
  unsigned* adj     = (unsigned*)(misc + 4096);                      // 32 KB
  int*      labels  = (int*)(misc + 4096 + K * 16 * 4);
  int*      compact = labels + K;
  float*    m2      = (float*)(compact + K);

  k_c2     <<<K, 256, 0, stream>>>(C, c2, maxbits);
  k_dmax   <<<(K * K) / 256, 256, 0, stream>>>(C, c2, maxbits);
  k_adj    <<<(K * 16) / 256, 256, 0, stream>>>(C, c2, maxbits, adj);
  k_closure<<<1, K, 0, stream>>>(adj, labels);
  k_compact<<<1, K, 0, stream>>>(labels, compact);
  k_merge  <<<(K * D) / 256, 256, 0, stream>>>(C, compact, merged);
  k_m2     <<<K, 256, 0, stream>>>(merged, m2);
  k_bsplit <<<(K * D) / 256, 256, 0, stream>>>(merged, mh, ml);
  k_assign <<<N / 128, 256, 0, stream>>>(X, mh, ml, m2, (int*)d_out);
}
